// LearableBlockOffset_26843545600773
// MI455X (gfx1250) — matrix operand order repaired and checked by reading, not compiled, not run on hardware
//
#include <hip/hip_runtime.h>
#include <hip/hip_bf16.h>

typedef __attribute__((ext_vector_type(16))) _Float16 v16h;
typedef __attribute__((ext_vector_type(8)))  _Float16 v8h;
typedef __attribute__((ext_vector_type(8)))  float    v8f;
typedef __attribute__((ext_vector_type(4)))  float    v4f;

#define D_CH   256
#define M_ROWS (32 * 2048)   // 65536 rows
#define HW     1024          // 32*32 spatial
#define WIMG   32
#define A_PAD  264           // LDS row stride in halfs: 132 dwords, 132%64=4 -> conflict-free

// ---------------------------------------------------------------------------
// Gather: x[row, ch] = gimage[b, ch, (py/8)*32 + (px/8)]
// ---------------------------------------------------------------------------
__global__ void k_gather(const float* __restrict__ gimage,
                         const int* __restrict__ pts,
                         float* __restrict__ X) {
  int row = blockIdx.x;          // b*2048 + cand
  int ch  = threadIdx.x;         // 0..255
  int b   = row >> 11;
  int iy  = pts[row * 2 + 0] >> 3;
  int ix  = pts[row * 2 + 1] >> 3;
  int lin = iy * WIMG + ix;
  X[(size_t)row * D_CH + ch] = gimage[((size_t)b * D_CH + ch) * HW + lin];
}

// ---------------------------------------------------------------------------
// Strided fp32 -> f16 weight convert (stride=9, offs=4 pulls conv center tap)
// ---------------------------------------------------------------------------
__global__ void k_cvt_w(const float* __restrict__ src, _Float16* __restrict__ dst,
                        int stride, int offs, int n) {
  int i = blockIdx.x * blockDim.x + threadIdx.x;
  if (i < n) dst[i] = (_Float16)src[(size_t)i * stride + offs];
}

// ---------------------------------------------------------------------------
// Fused layer kernel. Block = 256 threads = 8 waves = 32 rows x 256 cols.
//
// Phase 1 (prologue): each thread owns (row r = t>>3, 32 channels). Loads fp32,
//   MODE==1: GroupNorm (4 groups of 8 chans, thread-local stats) + SiLU,
//   converts to f16 into LDS A-tile [32][264].
//   CLS==1: also accumulates cls-logit partials from the raw fp32 row.
// Phase 2 (WMMA): wave w -> rows 16*(w>>2), cols 64*(w&3); A from LDS
//   (ds_load_b128, bank-conflict-free), B from L2-resident f16 weights,
//   8x K-steps of v_wmma_f32_16x16x32_f16 per output tile.
// Epilogue: bias (+skip) (+relu) -> Out;  BOX==1: stage ReLU'd tile in LDS,
//   reduce against mlp_w3, sigmoid, write boxes only (no H writeback).
// ---------------------------------------------------------------------------
template <int MODE, int CLS, int BOX>
__global__ __launch_bounds__(256) void k_fused(
    const float* __restrict__ X,
    const float* __restrict__ gnw, const float* __restrict__ gnb,
    const _Float16* __restrict__ W16,
    const float* __restrict__ bias,
    const float* __restrict__ skip,
    float* __restrict__ Out,
    int relu,
    const float* __restrict__ hw, const float* __restrict__ hb,
    float* __restrict__ hout) {

  __shared__ _Float16 Atile[32][A_PAD];
  __shared__ float cpart[(CLS ? 32 : 1)][8][2];

  const int t    = threadIdx.x;
  const int rloc = t >> 3;         // 0..31
  const int cseg = (t & 7) * 32;   // 0,32,...,224
  const int rowBlk = blockIdx.x * 32;

  // ---- Phase 1: load + (GN+SiLU) + f16 convert into LDS ----
  {
    const float* xr = X + (size_t)(rowBlk + rloc) * D_CH + cseg;
    float v[32];
#pragma unroll
    for (int q = 0; q < 8; ++q) {
      v4f tmp = *(const v4f*)(xr + q * 4);
#pragma unroll
      for (int j = 0; j < 4; ++j) v[q * 4 + j] = tmp[j];
    }
    if constexpr (CLS) {   // cls head uses raw fp32 activations
      float l0 = 0.f, l1 = 0.f;
#pragma unroll
      for (int j = 0; j < 32; ++j) {
        l0 += v[j] * hw[cseg + j];
        l1 += v[j] * hw[D_CH + cseg + j];
      }
      cpart[rloc][t & 7][0] = l0;
      cpart[rloc][t & 7][1] = l1;
    }
    if constexpr (MODE) {  // GroupNorm(8-ch groups) + SiLU
#pragma unroll
      for (int g = 0; g < 4; ++g) {
        float s = 0.f, sq = 0.f;
#pragma unroll
        for (int j = 0; j < 8; ++j) { float x = v[g * 8 + j]; s += x; sq += x * x; }
        float mu  = s * 0.125f;
        float var = sq * 0.125f - mu * mu;
        float rs  = rsqrtf(var + 1e-5f);
#pragma unroll
        for (int j = 0; j < 8; ++j) {
          int ch = cseg + g * 8 + j;
          float xn = (v[g * 8 + j] - mu) * rs * gnw[ch] + gnb[ch];
          v[g * 8 + j] = xn / (1.0f + expf(-xn));   // SiLU
        }
      }
    }
#pragma unroll
    for (int q = 0; q < 4; ++q) {
      v8h o;
#pragma unroll
      for (int j = 0; j < 8; ++j) o[j] = (_Float16)v[q * 8 + j];
      *(v8h*)(&Atile[rloc][cseg + q * 8]) = o;
    }
  }
  __syncthreads();

  if constexpr (CLS) {
    if (t < 64) {
      int r = t >> 1, o = t & 1;
      float s = 0.f;
#pragma unroll
      for (int j = 0; j < 8; ++j) s += cpart[r][j][o];
      hout[(size_t)(rowBlk + r) * 2 + o] = s + hb[o];
    }
  }

  // ---- Phase 2: WMMA GEMM (A from LDS, B from global/L2) ----
  const int w    = t >> 5;
  const int lane = t & 31;
  const int lm   = lane & 15;
  const int lh   = lane >> 4;
  const int mIdx = w >> 2;             // 0..1
  const int colBase = (w & 3) * 64;    // 0..192

  v8f acc[4];
#pragma unroll
  for (int tt = 0; tt < 4; ++tt)
#pragma unroll
    for (int e = 0; e < 8; ++e) acc[tt][e] = 0.0f;

  const _Float16* Ap = &Atile[mIdx * 16 + lm][lh * 16];

#pragma unroll
  for (int kt = 0; kt < 8; ++kt) {                 // K = 256
    v8h lo = *(const v8h*)(Ap + kt * 32);
    v8h hi = *(const v8h*)(Ap + kt * 32 + 8);
    v16h a;
#pragma unroll
    for (int i = 0; i < 8; ++i) { a[i] = lo[i]; a[8 + i] = hi[i]; }
#pragma unroll
    for (int tt = 0; tt < 4; ++tt) {
      const _Float16* Bp =
          W16 + (size_t)(colBase + tt * 16 + lm) * D_CH + kt * 32 + lh * 16;
      v16h b = *(const v16h*)Bp;
      acc[tt] = __builtin_amdgcn_wmma_f32_16x16x32_f16(
          false, a, false, b, (short)0, acc[tt], false, false);
    }
  }

  // ---- Epilogue ----
  if constexpr (!BOX) {
#pragma unroll
    for (int tt = 0; tt < 4; ++tt) {
      int cidx = colBase + tt * 16 + lm;
      float bb = bias[cidx];
#pragma unroll
      for (int vv = 0; vv < 8; ++vv) {
        int r = rowBlk + mIdx * 16 + lh * 8 + vv;
        float val = acc[tt][vv] + bb;
        if (skip) val += skip[(size_t)r * D_CH + cidx];
        if (relu) val = fmaxf(val, 0.0f);
        Out[(size_t)r * D_CH + cidx] = val;
      }
    }
  } else {
    // Stage ReLU'd H tile to LDS, reduce against mlp_w3, sigmoid -> boxes.
    __shared__ float Hs[32][257];
    __shared__ float bpart[32][8][4];
#pragma unroll
    for (int tt = 0; tt < 4; ++tt) {
      int cidx = colBase + tt * 16 + lm;
      float bb = bias[cidx];
#pragma unroll
      for (int vv = 0; vv < 8; ++vv) {
        int rl = mIdx * 16 + lh * 8 + vv;
        Hs[rl][cidx] = fmaxf(acc[tt][vv] + bb, 0.0f);
      }
    }
    __syncthreads();
    {
      int r = t >> 3, j = t & 7;
      float p0 = 0.f, p1 = 0.f, p2 = 0.f, p3 = 0.f;
#pragma unroll
      for (int c = 0; c < 32; ++c) {
        float hv = Hs[r][j * 32 + c];
        int ch = j * 32 + c;
        p0 += hv * hw[ch];
        p1 += hv * hw[D_CH + ch];
        p2 += hv * hw[2 * D_CH + ch];
        p3 += hv * hw[3 * D_CH + ch];
      }
      bpart[r][j][0] = p0; bpart[r][j][1] = p1;
      bpart[r][j][2] = p2; bpart[r][j][3] = p3;
    }
    __syncthreads();
    if (t < 128) {
      int r = t >> 2, o = t & 3;
      float s = 0.f;
#pragma unroll
      for (int j = 0; j < 8; ++j) s += bpart[r][j][o];
      hout[(size_t)(rowBlk + r) * 4 + o] = 1.0f / (1.0f + expf(-(s + hb[o])));
    }
  }
}

// ---------------------------------------------------------------------------
extern "C" void kernel_launch(void* const* d_in, const int* in_sizes, int n_in,
                              void* d_out, int out_size, void* d_ws, size_t ws_size,
                              hipStream_t stream) {
  const float* gimage = (const float*)d_in[0];
  const int*   pts    = (const int*)d_in[1];
  const float* gn1w   = (const float*)d_in[2];
  const float* gn1b   = (const float*)d_in[3];
  const float* c1w    = (const float*)d_in[4];
  const float* c1b    = (const float*)d_in[5];
  const float* gn2w   = (const float*)d_in[6];
  const float* gn2b   = (const float*)d_in[7];
  const float* c2w    = (const float*)d_in[8];
  const float* c2b    = (const float*)d_in[9];
  const float* clsw   = (const float*)d_in[10];
  const float* clsb   = (const float*)d_in[11];
  const float* m1w    = (const float*)d_in[12];
  const float* m1b    = (const float*)d_in[13];
  const float* m2w    = (const float*)d_in[14];
  const float* m2b    = (const float*)d_in[15];
  const float* m3w    = (const float*)d_in[16];
  const float* m3b    = (const float*)d_in[17];

  char* ws = (char*)d_ws;
  const size_t actF32 = (size_t)M_ROWS * D_CH * sizeof(float);   // 64 MB
  float*    bufX = (float*)ws;                                   // activations
  float*    bufH = (float*)(ws + actF32);                        // MLP hidden
  _Float16* W16  = (_Float16*)(ws + 2 * actF32);                 // f16 weights

  const int NW = D_CH * D_CH;  // 65536 elements per matrix
  _Float16* Wc1[2] = {W16,          W16 + 1 * NW};
  _Float16* Wc2[2] = {W16 + 2 * NW, W16 + 3 * NW};
  _Float16* Wm1    =  W16 + 4 * NW;
  _Float16* Wm2    =  W16 + 5 * NW;

  dim3 cg((NW + 255) / 256), cb(256);
  for (int i = 0; i < 2; ++i) {
    k_cvt_w<<<cg, cb, 0, stream>>>(c1w + (size_t)i * NW * 9, Wc1[i], 9, 4, NW);
    k_cvt_w<<<cg, cb, 0, stream>>>(c2w + (size_t)i * NW * 9, Wc2[i], 9, 4, NW);
  }
  k_cvt_w<<<cg, cb, 0, stream>>>(m1w, Wm1, 1, 0, NW);
  k_cvt_w<<<cg, cb, 0, stream>>>(m2w, Wm2, 1, 0, NW);

  k_gather<<<M_ROWS, 256, 0, stream>>>(gimage, pts, bufX);

  const int GB = M_ROWS / 32;   // 2048 blocks per fused layer
  float* outL = (float*)d_out;                 // [65536, 2]
  float* outB = outL + (size_t)M_ROWS * 2;     // [65536, 4]

  // Two residual blocks: GN+SiLU fused into each GEMM
  for (int i = 0; i < 2; ++i) {
    k_fused<1, 0, 0><<<GB, 256, 0, stream>>>(
        bufX, gn1w + i * D_CH, gn1b + i * D_CH, Wc1[i], c1b + i * D_CH,
        nullptr, bufH, 0, nullptr, nullptr, nullptr);
    k_fused<1, 0, 0><<<GB, 256, 0, stream>>>(
        bufH, gn2w + i * D_CH, gn2b + i * D_CH, Wc2[i], c2b + i * D_CH,
        /*skip=*/bufX, bufX, 0, nullptr, nullptr, nullptr);
  }

  // MLP layer 1 + fused cls-logits head (logits read pre-MLP activations)
  k_fused<0, 1, 0><<<GB, 256, 0, stream>>>(
      bufX, nullptr, nullptr, Wm1, m1b, nullptr, bufH, 1, clsw, clsb, outL);

  // MLP layer 2 + fused box head (no H writeback at all)
  k_fused<0, 0, 1><<<GB, 256, 0, stream>>>(
      bufH, nullptr, nullptr, Wm2, m2b, nullptr, nullptr, 1, m3w, m3b, outB);
}
